// GroupedQueryAttention_6511170421705
// MI455X (gfx1250) — compile-verified, matrix operand order checked by reading, not run on hardware
//
#include <hip/hip_runtime.h>
#include <hip/hip_bf16.h>

// ---------------------------------------------------------------------------
// GQA (32 heads / 8 KV groups / head_dim 64, seq 2048, dmodel 2048) for
// MI455X (gfx1250, wave32). Matmuls: v_wmma_f32_16x16x32_bf16.
// GEMMs are TDM-pipelined: tensor_load_to_lds double-buffers K-slabs into LDS
// while WMMAs consume the previous slab (TENSORcnt-tracked async DMA).
// ---------------------------------------------------------------------------

#define SEQ   2048
#define DIN   2048
#define DOUT  2048
#define NHEADS 32
#define GROUPS 8
#define HDIM   64
#define KVDIM  (GROUPS * HDIM)   // 512
#define KSLAB  64                // K-depth staged per TDM slab

typedef __attribute__((ext_vector_type(16))) __bf16 v16bf;
typedef __attribute__((ext_vector_type(8)))  float  v8f;
typedef __attribute__((ext_vector_type(4)))  unsigned int v4ui;
typedef __attribute__((ext_vector_type(8)))  int    v8si;
typedef __attribute__((ext_vector_type(4)))  int    v4si;

#if defined(__has_builtin)
#if __has_builtin(__builtin_amdgcn_tensor_load_to_lds) && \
    __has_builtin(__builtin_amdgcn_s_wait_tensorcnt)
#define HAVE_TDM 1
#endif
#endif
#ifndef HAVE_TDM
#define HAVE_TDM 0
#endif

union Frag {
    v16bf v;
    uint4 u[2];
};

static __device__ inline v8f v8f_zero() {
    v8f z;
#pragma unroll
    for (int i = 0; i < 8; ++i) z[i] = 0.0f;
    return z;
}

static __device__ inline v8f wmma_bf16(const Frag& a, const Frag& b, v8f c) {
    // (neg_a, A, neg_b, B, c_mod, C, reuse_a, reuse_b)
    return __builtin_amdgcn_wmma_f32_16x16x32_bf16(false, a.v, false, b.v,
                                                   (short)0, c, false, false);
}

static __device__ inline Frag ld_frag(const __bf16* p) {
    Frag f;
    f.u[0] = *(const uint4*)(p);
    f.u[1] = *(const uint4*)(p + 16);
    return f;
}

#if HAVE_TDM
// ---------------------------------------------------------------------------
// TDM: async 2D tile DMA (global bf16 -> LDS), descriptor per ISA ch.8.
//   tensor_w/h : remaining tensor extent from tile start (elements / rows)
//   tile_w/h   : tile extent (elements / rows); tile packed row-major in LDS
//   row_stride : global row pitch in elements
// ---------------------------------------------------------------------------
static __device__ inline void tdm_load_2d(unsigned lds_byte_addr, const void* gaddr,
                                          unsigned tensor_w, unsigned tensor_h,
                                          unsigned tile_w, unsigned tile_h,
                                          unsigned row_stride) {
    unsigned long long ga = (unsigned long long)(uintptr_t)gaddr;
    v4ui g0;
    g0[0] = 1u;                                          // count=1, user descriptor
    g0[1] = lds_byte_addr;                               // lds_addr (bytes)
    g0[2] = (unsigned)(ga & 0xffffffffu);                // global_addr[31:0]
    g0[3] = (unsigned)((ga >> 32) & 0x01ffffffu)         // global_addr[56:32]
          | (2u << 30);                                  // type = 2 ("image")
    v8si g1;
    g1[0] = (int)(1u << 16);                             // data_size=1 (2B), mask=0
    g1[1] = (int)((tensor_w & 0xffffu) << 16);           // tensor_dim0[15:0]
    g1[2] = (int)((tensor_w >> 16) | ((tensor_h & 0xffffu) << 16)); // dim0 hi|dim1 lo
    g1[3] = (int)((tensor_h >> 16) | (tile_w << 16));    // dim1 hi | tile_dim0
    g1[4] = (int)(tile_h & 0xffffu);                     // tile_dim1 ; tile_dim2=0
    g1[5] = (int)row_stride;                             // tensor_dim0_stride[31:0]
    g1[6] = 0;                                           // stride hi | dim1_stride lo
    g1[7] = 0;
    v4si z4 = {0, 0, 0, 0};
#if __clang_major__ >= 23
    v8si z8 = {0, 0, 0, 0, 0, 0, 0, 0};
    __builtin_amdgcn_tensor_load_to_lds(g0, g1, z4, z4, z8, 0);
#else
    __builtin_amdgcn_tensor_load_to_lds(g0, g1, z4, z4, 0);
#endif
}
#endif

// ---------------------------------------------------------------------------
// f32 -> bf16 elementwise convert
// ---------------------------------------------------------------------------
__global__ void cvt_bf16(const float* __restrict__ in, __bf16* __restrict__ out, int n) {
    int i = blockIdx.x * blockDim.x + threadIdx.x;
    if (i < n) out[i] = (__bf16)in[i];
}

// in: [K][N] f32 row-major  ->  out: [N][K] bf16 row-major (transpose)
__global__ void cvt_T_bf16(const float* __restrict__ in, __bf16* __restrict__ out,
                           int K, int N) {
    int i = blockIdx.x * blockDim.x + threadIdx.x;
    if (i < K * N) {
        int k = i % K;
        int n = i / K;
        out[(size_t)n * K + k] = (__bf16)in[(size_t)k * N + n];
    }
}

// ---------------------------------------------------------------------------
// GEMM: C[M][N] = A[M][K] * Bt[N][K]^T   (A,Bt bf16 row-major; C f32 or bf16)
// One wave per 32x64 C tile (8 accumulator fragments).
// TDM path: double-buffered K-slabs (A 32xKSLAB, B 64xKSLAB) staged to LDS by
// the Tensor Data Mover; slab s+1 DMA overlaps WMMAs on slab s.
//
// Fragment layouts per CDNA5 ISA 7.12.2 (16-bit A 16x32):
//   lane = khalf*16 + m ; per lane two contiguous 16B reads at K offsets
//   khalf*8 and 16+khalf*8. B (rows of Bt) identical with N in place of M.
// C/D: lane = n (0..15 twice); VGPR r holds row m = r + (lane>=16 ? 8 : 0).
// ---------------------------------------------------------------------------
template <bool STORE_BF16>
__global__ __launch_bounds__(32)
void gemm_bf16_wmma(const __bf16* __restrict__ A, const __bf16* __restrict__ Bt,
                    void* __restrict__ Cout, int M, int N, int K) {
    const int lane  = threadIdx.x & 31;
    const int r16   = lane & 15;
    const int khalf = lane >> 4;
    const int m0 = blockIdx.x * 32;
    const int n0 = blockIdx.y * 64;

    v8f acc[2][4];
#pragma unroll
    for (int mt = 0; mt < 2; ++mt)
#pragma unroll
        for (int t = 0; t < 4; ++t) acc[mt][t] = v8f_zero();

#if HAVE_TDM
    __shared__ __align__(16) __bf16 sA[2][32 * KSLAB];
    __shared__ __align__(16) __bf16 sB[2][64 * KSLAB];

    const int nslab = K / KSLAB;
    // -- issue slab 0 (A then B; TDM ignores EXEC, one issue per wave) --
    tdm_load_2d((unsigned)(uintptr_t)&sA[0][0], A + (size_t)m0 * K,
                (unsigned)K, (unsigned)(M - m0), KSLAB, 32, (unsigned)K);
    tdm_load_2d((unsigned)(uintptr_t)&sB[0][0], Bt + (size_t)n0 * K,
                (unsigned)K, (unsigned)(N - n0), KSLAB, 64, (unsigned)K);

    for (int s = 0; s < nslab; ++s) {
        const int cur = s & 1;
        if (s + 1 < nslab) {
            const int k1 = (s + 1) * KSLAB;
            // ensure prior ds reads of the buffer being overwritten retired
            asm volatile("s_wait_dscnt 0" ::: "memory");
            tdm_load_2d((unsigned)(uintptr_t)&sA[cur ^ 1][0],
                        A + (size_t)m0 * K + k1,
                        (unsigned)(K - k1), (unsigned)(M - m0), KSLAB, 32, (unsigned)K);
            tdm_load_2d((unsigned)(uintptr_t)&sB[cur ^ 1][0],
                        Bt + (size_t)n0 * K + k1,
                        (unsigned)(K - k1), (unsigned)(N - n0), KSLAB, 64, (unsigned)K);
            __builtin_amdgcn_s_wait_tensorcnt((short)2);   // slab s complete
        } else {
            __builtin_amdgcn_s_wait_tensorcnt((short)0);
        }
#pragma unroll
        for (int kkloc = 0; kkloc < KSLAB; kkloc += 32) {
            Frag aF[2];
#pragma unroll
            for (int mt = 0; mt < 2; ++mt)
                aF[mt] = ld_frag(&sA[cur][(mt * 16 + r16) * KSLAB + kkloc + khalf * 8]);
#pragma unroll
            for (int t = 0; t < 4; ++t) {
                Frag bF = ld_frag(&sB[cur][(t * 16 + r16) * KSLAB + kkloc + khalf * 8]);
                acc[0][t] = wmma_bf16(aF[0], bF, acc[0][t]);
                acc[1][t] = wmma_bf16(aF[1], bF, acc[1][t]);
            }
        }
    }
#else
    // Fallback: direct global->register fragments (no TDM builtin available).
    const __bf16* aptr = A + (size_t)(m0 + r16) * K + khalf * 8;
    for (int kk = 0; kk < K; kk += 32) {
        Frag aF[2];
#pragma unroll
        for (int mt = 0; mt < 2; ++mt)
            aF[mt] = ld_frag(aptr + (size_t)mt * 16 * K + kk);
#pragma unroll
        for (int t = 0; t < 4; ++t) {
            Frag bF = ld_frag(Bt + (size_t)(n0 + t * 16 + r16) * K + khalf * 8 + kk);
            acc[0][t] = wmma_bf16(aF[0], bF, acc[0][t]);
            acc[1][t] = wmma_bf16(aF[1], bF, acc[1][t]);
        }
    }
#endif

    const int rbase = khalf * 8;
#pragma unroll
    for (int mt = 0; mt < 2; ++mt)
#pragma unroll
        for (int t = 0; t < 4; ++t)
#pragma unroll
            for (int r = 0; r < 8; ++r) {
                size_t off = (size_t)(m0 + mt * 16 + rbase + r) * N + (n0 + t * 16 + r16);
                if (STORE_BF16) ((__bf16*)Cout)[off] = (__bf16)acc[mt][t][r];
                else            ((float*)Cout)[off]  = acc[mt][t][r];
            }
}

// ---------------------------------------------------------------------------
// Flash-style causal attention. One wave per (16-query tile, head).
//   Qb  : [SEQ][DOUT]  bf16 (head h at col h*64)
//   Kb  : [SEQ][KVDIM] bf16 (group g at col g*64; rows act as Bt)
//   Vtb : [KVDIM][SEQ] bf16 (V transposed; rows act as Bt)
//   ctxb: [SEQ][DOUT]  bf16 out
// ---------------------------------------------------------------------------
__global__ __launch_bounds__(32)
void attn_kernel(const __bf16* __restrict__ Qb, const __bf16* __restrict__ Kb,
                 const __bf16* __restrict__ Vtb, __bf16* __restrict__ ctxb) {
    const int lane  = threadIdx.x & 31;
    const int r16   = lane & 15;
    const int khalf = lane >> 4;
    const int rbase = khalf * 8;
    const int qt = blockIdx.x;
    const int h  = blockIdx.y;
    const int g  = h >> 2;           // heads per group = 4
    const int q0 = qt * 16;

    __shared__ float  s_s[16 * 32];
    __shared__ __align__(16) __bf16 s_p[16 * 32];
    __shared__ float  s_a[16];
    __shared__ float  s_r[16];

    Frag qf[2];
    {
        const __bf16* qp = Qb + (size_t)(q0 + r16) * DOUT + h * HDIM + khalf * 8;
        qf[0] = ld_frag(qp);
        qf[1] = ld_frag(qp + 32);
    }

    v8f o[4];
#pragma unroll
    for (int t = 0; t < 4; ++t) o[t] = v8f_zero();
    float m_run = -INFINITY;
    float lsum  = 0.0f;

    const float LOG2E = 1.44269504088896f;
    const int nblk = (q0 + 16 + 31) >> 5;   // causal: keys 0 .. q0+15

    for (int blk = 0; blk < nblk; ++blk) {
        const int kb = blk * 32;

        v8f s[2];
#pragma unroll
        for (int t = 0; t < 2; ++t) {
            const __bf16* kp = Kb + (size_t)(kb + t * 16 + r16) * KVDIM + g * HDIM + khalf * 8;
            Frag k0 = ld_frag(kp);
            Frag k1 = ld_frag(kp + 32);
            v8f a = v8f_zero();
            a = wmma_bf16(qf[0], k0, a);
            a = wmma_bf16(qf[1], k1, a);
            s[t] = a;
        }

#pragma unroll
        for (int t = 0; t < 2; ++t) {
#pragma unroll
            for (int r = 0; r < 8; ++r) {
                int q    = q0 + rbase + r;
                int kidx = kb + t * 16 + r16;
                float v  = s[t][r] * 0.125f;           // 1/sqrt(64)
                if (kidx > q) v = -INFINITY;
                s_s[(rbase + r) * 32 + t * 16 + r16] = v;
            }
        }
        __syncthreads();

        if (lane < 16) {
            float mb = -INFINITY;
            for (int j = 0; j < 32; ++j) mb = fmaxf(mb, s_s[lane * 32 + j]);
            float mn    = fmaxf(m_run, mb);
            float alpha = exp2f((m_run - mn) * LOG2E);
            float ps = 0.0f;
            for (int j = 0; j < 32; ++j) {
                float p = exp2f((s_s[lane * 32 + j] - mn) * LOG2E);
                ps += p;
                s_p[lane * 32 + j] = (__bf16)p;
            }
            lsum  = lsum * alpha + ps;
            m_run = mn;
            s_a[lane] = alpha;
        }
        __syncthreads();

#pragma unroll
        for (int t = 0; t < 4; ++t)
#pragma unroll
            for (int r = 0; r < 8; ++r) o[t][r] *= s_a[rbase + r];

        Frag pf = ld_frag(&s_p[r16 * 32 + khalf * 8]);
#pragma unroll
        for (int t = 0; t < 4; ++t) {
            Frag vf = ld_frag(Vtb + (size_t)(g * HDIM + t * 16 + r16) * SEQ + kb + khalf * 8);
            o[t] = wmma_bf16(pf, vf, o[t]);
        }
        __syncthreads();
    }

    if (lane < 16) s_r[lane] = 1.0f / lsum;
    __syncthreads();
#pragma unroll
    for (int t = 0; t < 4; ++t)
#pragma unroll
        for (int r = 0; r < 8; ++r)
            ctxb[(size_t)(q0 + rbase + r) * DOUT + h * HDIM + t * 16 + r16] =
                (__bf16)(o[t][r] * s_r[rbase + r]);
}

// ---------------------------------------------------------------------------
// Host-side orchestration
// ---------------------------------------------------------------------------
extern "C" void kernel_launch(void* const* d_in, const int* in_sizes, int n_in,
                              void* d_out, int out_size, void* d_ws, size_t ws_size,
                              hipStream_t stream) {
    const float* x  = (const float*)d_in[0];   // [1,2048,2048]
    const float* wq = (const float*)d_in[1];   // [2048,2048]
    const float* wk = (const float*)d_in[2];   // [2048,512]
    const float* wv = (const float*)d_in[3];   // [2048,512]
    const float* wo = (const float*)d_in[4];   // [2048,2048]
    float* out = (float*)d_out;

    char* ws = (char*)d_ws;
    size_t off = 0;
    auto alloc = [&](size_t bytes) -> void* {
        void* p = ws + off;
        off += (bytes + 255) & ~(size_t)255;
        return p;
    };

    __bf16* xb   = (__bf16*)alloc((size_t)SEQ * DIN * 2);
    __bf16* wqT  = (__bf16*)alloc((size_t)DOUT * DIN * 2);
    __bf16* wkT  = (__bf16*)alloc((size_t)KVDIM * DIN * 2);
    __bf16* wvT  = (__bf16*)alloc((size_t)KVDIM * DIN * 2);
    __bf16* woT  = (__bf16*)alloc((size_t)DOUT * DOUT * 2);
    __bf16* Qb   = (__bf16*)alloc((size_t)SEQ * DOUT * 2);
    __bf16* Kbuf = (__bf16*)alloc((size_t)SEQ * KVDIM * 2);
    __bf16* Vtb  = (__bf16*)alloc((size_t)KVDIM * SEQ * 2);
    __bf16* ctxb = (__bf16*)alloc((size_t)SEQ * DOUT * 2);

    const int T = 256;
    cvt_bf16<<<(SEQ * DIN + T - 1) / T, T, 0, stream>>>(x, xb, SEQ * DIN);
    cvt_T_bf16<<<(DIN * DOUT + T - 1) / T, T, 0, stream>>>(wq, wqT, DIN, DOUT);
    cvt_T_bf16<<<(DIN * KVDIM + T - 1) / T, T, 0, stream>>>(wk, wkT, DIN, KVDIM);
    cvt_T_bf16<<<(DIN * KVDIM + T - 1) / T, T, 0, stream>>>(wv, wvT, DIN, KVDIM);
    cvt_T_bf16<<<(DOUT * DOUT + T - 1) / T, T, 0, stream>>>(wo, woT, DOUT, DOUT);

    // Q = x @ wq                         [2048 x 2048]
    gemm_bf16_wmma<true><<<dim3(SEQ / 32, DOUT / 64), 32, 0, stream>>>(
        xb, wqT, Qb, SEQ, DOUT, DIN);
    // K = x @ wk                         [2048 x 512]
    gemm_bf16_wmma<true><<<dim3(SEQ / 32, KVDIM / 64), 32, 0, stream>>>(
        xb, wkT, Kbuf, SEQ, KVDIM, DIN);
    // V^T = wv^T @ x^T  (Bt := xb)       [512 x 2048]
    gemm_bf16_wmma<true><<<dim3(KVDIM / 32, SEQ / 64), 32, 0, stream>>>(
        wvT, xb, Vtb, KVDIM, SEQ, DIN);

    // causal GQA attention -> ctx (bf16)
    attn_kernel<<<dim3(SEQ / 16, NHEADS), 32, 0, stream>>>(Qb, Kbuf, Vtb, ctxb);

    // out = ctx @ wo (f32 result)
    gemm_bf16_wmma<false><<<dim3(SEQ / 32, DOUT / 64), 32, 0, stream>>>(
        ctxb, woT, out, SEQ, DOUT, DOUT);
}
